// CustomLSTM_1846835937437
// MI455X (gfx1250) — matrix operand order repaired and checked by reading, not compiled, not run on hardware
//
#include <hip/hip_runtime.h>

// ---------------- problem constants ----------------
#define SEQ_  1024
#define B_    64
#define IN_   512
#define H_    512
#define K_    1024          // IN + H
#define NWG_  16            // workgroups per layer
#define HC_   32            // hidden units owned per WG
#define GR_   128           // gate rows per WG (4 * HC_)
#define RING_ 8             // h0 ring depth (steps)
#define THREADS_ 256

// ---------------- LDS layout (bytes) ----------------
#define WSTR_ 1024          // bf16 elems per weight row (no pad; XOR block swizzle)
#define ASTR_ 136           // bf16 elems per A-chunk row (128 data + 4-dword TDM pad)
#define ZSTR_ 128           // f32 elems per Z row
#define KC_   128           // K-chunk (4 WMMA k-steps)
#define NKC_  8             // chunks per step (K=1024)

#define WOFF_ 0
#define ABYTES1_ (B_ * ASTR_ * 2)                  // 17408
#define AOFF_ (WOFF_ + GR_ * WSTR_ * 2)            // 262144
#define ZOFF_ AOFF_                                // zbuf aliases abuf (disjoint lifetimes)
#define SMEM_ (AOFF_ + 2 * ABYTES1_)               // 296960  (< 320 KB WGP LDS)

// ---------------- workspace layout (bytes) ----------------
#define WS_FLAGS_   0                                   // flag0[SEQ], flag1[SEQ] u32
#define WS_ZERO_    8192                                // 64x512 bf16 zeros (t=0 h source)
#define WS_RING0_   (WS_ZERO_ + B_ * H_ * 2)            // [8][B][H] bf16 h0
#define WS_RING1_   (WS_RING0_ + RING_ * B_ * H_ * 2)   // [2][B][H] bf16 h1
#define WS_XBF_     (WS_RING1_ + 2 * B_ * H_ * 2)       // [SEQ][B][IN] bf16 x
#define WS_ZERO_WORDS_ ((WS_RING0_ - WS_FLAGS_) / 4)    // flags + zerobuf zeroed

typedef __attribute__((ext_vector_type(16))) __bf16 v16bf;
typedef __attribute__((ext_vector_type(8)))  __bf16 v8bf;
typedef __attribute__((ext_vector_type(8)))  float  v8f;
typedef __attribute__((ext_vector_type(4)))  unsigned int u32x4;
typedef __attribute__((ext_vector_type(4)))  int          i32x4;
typedef __attribute__((ext_vector_type(8)))  int          i32x8;

#if defined(__has_builtin)
#if __has_builtin(__builtin_amdgcn_tensor_load_to_lds) && __has_builtin(__builtin_amdgcn_s_wait_tensorcnt)
#define USE_TDM 1
#endif
#endif

// float -> bf16 (round-to-nearest-even), as raw u16
__device__ __forceinline__ unsigned f2bf(float f) {
    unsigned u = __float_as_uint(f);
    return (u + 0x7FFFu + ((u >> 16) & 1u)) >> 16;
}
__device__ __forceinline__ unsigned pack2bf(float a, float b) {
    return f2bf(a) | (f2bf(b) << 16);
}
__device__ __forceinline__ float sigmoidf_(float x) {
    return 1.0f / (1.0f + __expf(-x));
}
__device__ __forceinline__ float tanh_fast(float x) {
    return 1.0f - 2.0f / (1.0f + __expf(2.0f * x));   // exact limits at +-inf, no NaN
}

__device__ __forceinline__ void wait_flag(const unsigned* f, unsigned target) {
    if (threadIdx.x == 0) {
        while (__hip_atomic_load(f, __ATOMIC_ACQUIRE, __HIP_MEMORY_SCOPE_AGENT) < target) {
            __builtin_amdgcn_s_sleep(1);
        }
    }
    __syncthreads();
}

__device__ __forceinline__ void signal_flag(unsigned* f) {
    __threadfence();
    __syncthreads();
    if (threadIdx.x == 0) {
        __hip_atomic_fetch_add(f, 1u, __ATOMIC_RELEASE, __HIP_MEMORY_SCOPE_AGENT);
    }
}

#ifdef USE_TDM
// Issue one TDM 2D tile copy: 64 rows x 128 bf16 cols, tensor row stride 512 elems,
// LDS dest padded 4 dwords after each 64-dword row (-> 136-elem padded stride).
__device__ __forceinline__ void tdm_issue(const unsigned short* src_elem,
                                          unsigned lds_byte_addr) {
    const unsigned long long ga = (unsigned long long)(size_t)src_elem;
    const unsigned w0g0 = 1u;                                 // count=1, user mode
    const unsigned w3g0 = (unsigned)(ga >> 32) | 0x80000000u; // addr[56:32] | type=2
    u32x4 g0 = { w0g0, lds_byte_addr, (unsigned)(ga & 0xffffffffu), w3g0 };

    // group1: data_size=2B, pad_enable, pad_interval=64 dwords (code 5),
    // pad_amount=4 dwords (code 3); tensor_dim0=512, tensor_dim1=64,
    // tile_dim0=128, tile_dim1=64, tensor_dim0_stride=512
    const int g1w0 = (1 << 16) | (1 << 20) | (5 << 22) | (3 << 25);
    const int g1w1 = (512 & 0xffff) << 16;     // tensor_dim0[15:0]
    const int g1w2 = (64 << 16);               // tensor_dim0[31:16]=0 | tensor_dim1[15:0]
    const int g1w3 = (128 << 16);              // tensor_dim1[31:16]=0 | tile_dim0
    const int g1w4 = 64;                       // tile_dim1 | tile_dim2=0
    const int g1w5 = 512;                      // tensor_dim0_stride[31:0]
    i32x8 g1 = { g1w0, g1w1, g1w2, g1w3, g1w4, g1w5, 0, 0 };
    i32x4 g2 = { 0, 0, 0, 0 };
    i32x4 g3 = { 0, 0, 0, 0 };
#if __clang_major__ >= 23
    i32x8 g4 = { 0, 0, 0, 0, 0, 0, 0, 0 };
    __builtin_amdgcn_tensor_load_to_lds(g0, g1, g2, g3, g4, 0);
#else
    __builtin_amdgcn_tensor_load_to_lds(g0, g1, g2, g3, 0);
#endif
}
#endif

// weight fragment pointer with XOR block swizzle (32B granules, 2-way max bank spread)
__device__ __forceinline__ const v16bf* bfrag_ptr(const unsigned short* wbuf, int n, int c) {
    const int blk = (c >> 4) ^ (n & 7);
    return (const v16bf*)(wbuf + n * WSTR_ + (blk << 4));
}

// load A fragment + 4 B fragments for one 16x16x32 k-step
__device__ __forceinline__ void load_frags(v16bf& a, v16bf (&b)[4],
                                           const unsigned short* ab,
                                           const unsigned short* wbuf,
                                           int mtile, int nhalf, int lh, int hi,
                                           int kabs, int kloc) {
    const unsigned short* ap = ab + (mtile * 16 + lh) * ASTR_ + kloc + hi * 16;
    v8bf alo = *(const v8bf*)(ap);
    v8bf ahi = *(const v8bf*)(ap + 8);
    a = __builtin_shufflevector(alo, ahi,
        0, 1, 2, 3, 4, 5, 6, 7, 8, 9, 10, 11, 12, 13, 14, 15);
    const int n0 = nhalf * 64 + lh;
    const int c  = kabs + hi * 16;
#pragma unroll
    for (int j = 0; j < 4; ++j)
        b[j] = *bfrag_ptr(wbuf, n0 + j * 16, c);
}

__global__ void __launch_bounds__(256, 1) ws_init(unsigned* p, int n) {
    int i = blockIdx.x * blockDim.x + threadIdx.x;
    if (i < n) p[i] = 0u;
}

// one-time fp32 -> bf16 conversion of the whole input sequence
__global__ void __launch_bounds__(256, 1) xcast(const float* __restrict__ src,
                                                unsigned short* __restrict__ dst) {
    const size_t i = ((size_t)blockIdx.x * blockDim.x + threadIdx.x) * 8;
    const float4 a = *(const float4*)(src + i);
    const float4 b = *(const float4*)(src + i + 4);
    uint4 o;
    o.x = pack2bf(a.x, a.y); o.y = pack2bf(a.z, a.w);
    o.z = pack2bf(b.x, b.y); o.w = pack2bf(b.z, b.w);
    *(uint4*)(dst + i) = o;
}

__global__ void __launch_bounds__(THREADS_, 1)
lstm_persistent(const unsigned short* __restrict__ xbf,
                const float* __restrict__ W0, const float* __restrict__ b0,
                const float* __restrict__ W1, const float* __restrict__ b1,
                float* __restrict__ out,
                unsigned short* __restrict__ ring0,
                unsigned short* __restrict__ ring1,
                const unsigned short* __restrict__ zerobuf,
                unsigned* __restrict__ flag0,
                unsigned* __restrict__ flag1) {
    extern __shared__ unsigned char smem[];
    unsigned short* wbuf = (unsigned short*)(smem + WOFF_);   // [GR_][WSTR_] bf16 weights (swizzled)
    unsigned short* abuf = (unsigned short*)(smem + AOFF_);   // [2][B_][ASTR_] bf16 A chunks
    float*          zbuf = (float*)(smem + ZOFF_);            // [B_][ZSTR_]  (aliases abuf)

    const int tid   = threadIdx.x;
    const int layer = blockIdx.x >> 4;
    const int hc    = blockIdx.x & 15;
    const int hcol0 = hc * HC_;

    const float* Wl = layer ? W1 : W0;
    const float* bl = layer ? b1 : b0;

    // ---- one-time weight preload: fp32 -> bf16, XOR block swizzle ----
    for (int idx = tid; idx < GR_ * (K_ / 4); idx += THREADS_) {
        const int r  = idx >> 8;          // 0..127
        const int c4 = idx & 255;         // float4 col group
        const int grow = (r >> 5) * H_ + hcol0 + (r & 31);
        const float4 w = *(const float4*)(Wl + (size_t)grow * K_ + c4 * 4);
        const int blk = (c4 >> 2) ^ (r & 7);
        unsigned* dst = (unsigned*)(wbuf + r * WSTR_ + (blk << 4) + ((c4 & 3) << 2));
        dst[0] = pack2bf(w.x, w.y);
        dst[1] = pack2bf(w.z, w.w);
    }

    // ---- per-wave tile assignment ----
    const int lane  = tid & 31;
    const int wv    = tid >> 5;
    const int mtile = wv >> 1;
    const int nhalf = wv & 1;
    const int lh    = lane & 15;
    const int hi    = lane >> 4;

    const int srow = tid >> 2;            // fallback staging row
    const int scol = (tid & 3) * 32;      // fallback staging 32-col group

    // per-thread gate biases (hh = tid&31 is constant across the whole sequence)
    const int hh = tid & 31;
    const float bi_ = bl[0 * H_ + hcol0 + hh];
    const float bf_ = bl[1 * H_ + hcol0 + hh];
    const float bo_ = bl[2 * H_ + hcol0 + hh];
    const float bg_ = bl[3 * H_ + hcol0 + hh];

    float creg[8];
#pragma unroll
    for (int q = 0; q < 8; ++q) creg[q] = 0.0f;

    const size_t FINAL_H = (size_t)SEQ_ * B_ * H_;
    const size_t FINAL_C = FINAL_H + (size_t)2 * B_ * H_;

    __syncthreads();

    for (int t = 0; t < SEQ_; ++t) {
        if (layer == 0) {
            if (t > 0)      wait_flag(flag0 + (t - 1), NWG_);
            if (t >= RING_) wait_flag(flag1 + (t - RING_), NWG_);
        } else {
            wait_flag(flag0 + t, NWG_);
            if (t > 0) wait_flag(flag1 + (t - 1), NWG_);
        }

        // A sources (bf16, row stride 512): cols [0,512) srcA, [512,1024) srcB
        const unsigned short* srcA;
        const unsigned short* srcB;
        if (layer == 0) {
            srcA = xbf + (size_t)t * B_ * IN_;
            srcB = (t > 0) ? (ring0 + (size_t)((t - 1) & (RING_ - 1)) * B_ * H_) : zerobuf;
            if (t + 1 < SEQ_)
                __builtin_prefetch(xbf + (size_t)(t + 1) * B_ * IN_ + (size_t)tid * 128, 0, 0);
        } else {
            srcA = ring0 + (size_t)(t & (RING_ - 1)) * B_ * H_;
            srcB = (t > 0) ? (ring1 + (size_t)((t - 1) & 1) * B_ * H_) : zerobuf;
        }

        v8f acc[4];
#pragma unroll
        for (int j = 0; j < 4; ++j) acc[j] = (v8f)(0.0f);

        // ---- stage chunk 0 ----
#ifdef USE_TDM
        if (wv == 0) {
            tdm_issue(srcA, (unsigned)(size_t)abuf);
            __builtin_amdgcn_s_wait_tensorcnt(0);
        }
#else
        {
            const unsigned short* p = srcA + (size_t)srow * 512 + scol;
            uint4 r0 = *(const uint4*)(p);
            uint4 r1 = *(const uint4*)(p + 8);
            uint4 r2 = *(const uint4*)(p + 16);
            uint4 r3 = *(const uint4*)(p + 24);
            unsigned short* d = abuf + srow * ASTR_ + scol;
            *(uint4*)(d)      = r0;
            *(uint4*)(d + 8)  = r1;
            *(uint4*)(d + 16) = r2;
            *(uint4*)(d + 24) = r3;
        }
#endif
        __syncthreads();

        // ---- K loop: 8 chunks of 128, TDM double buffered; within a chunk,
        //      fragments are register-double-buffered one k-step ahead ----
        for (int kc = 0; kc < NKC_; ++kc) {
            const bool have_next = (kc + 1 < NKC_);
#ifdef USE_TDM
            if (have_next && wv == 0) {
                const int kn = kc + 1;
                const unsigned short* src = (kn < 4) ? srcA : srcB;
                const int colbase = (kn < 4) ? kn * KC_ : (kn - 4) * KC_;
                tdm_issue(src + colbase,
                          (unsigned)(size_t)(abuf + ((kc + 1) & 1) * (B_ * ASTR_)));
            }
#else
            uint4 nx0, nx1, nx2, nx3;
            if (have_next) {
                const int kn = kc + 1;
                const unsigned short* src = (kn < 4) ? srcA : srcB;
                const int colbase = (kn < 4) ? kn * KC_ : (kn - 4) * KC_;
                const unsigned short* p = src + (size_t)srow * 512 + colbase + scol;
                nx0 = *(const uint4*)(p);
                nx1 = *(const uint4*)(p + 8);
                nx2 = *(const uint4*)(p + 16);
                nx3 = *(const uint4*)(p + 24);
            }
#endif

            // ---- WMMA on current chunk: 4 k-steps, frags prefetched 1 step ahead ----
            const unsigned short* ab = abuf + (kc & 1) * (B_ * ASTR_);
            const int k0 = kc * KC_;
            v16bf aF[2];
            v16bf bF[2][4];
            load_frags(aF[0], bF[0], ab, wbuf, mtile, nhalf, lh, hi, k0, 0);
#pragma unroll
            for (int ks = 0; ks < 4; ++ks) {
                const int cur = ks & 1;
                if (ks < 3)
                    load_frags(aF[cur ^ 1], bF[cur ^ 1], ab, wbuf, mtile, nhalf, lh, hi,
                               k0 + (ks + 1) * 32, (ks + 1) * 32);
#pragma unroll
                for (int j = 0; j < 4; ++j)
                    acc[j] = __builtin_amdgcn_wmma_f32_16x16x32_bf16(
                        false, aF[cur], false, bF[cur][j], (short)0, acc[j], false, false);
            }

#ifdef USE_TDM
            if (have_next && wv == 0) {
                __builtin_amdgcn_s_wait_tensorcnt(0);
            }
#else
            if (have_next) {
                unsigned short* d = abuf + ((kc + 1) & 1) * (B_ * ASTR_) +
                                    srow * ASTR_ + scol;
                *(uint4*)(d)      = nx0;
                *(uint4*)(d + 8)  = nx1;
                *(uint4*)(d + 16) = nx2;
                *(uint4*)(d + 24) = nx3;
            }
#endif
            __syncthreads();
        }

        // ---- spill Z accumulators to LDS (abuf is dead now; zbuf aliases it) ----
#pragma unroll
        for (int j = 0; j < 4; ++j) {
            const int n = nhalf * 64 + j * 16 + lh;
#pragma unroll
            for (int r = 0; r < 8; ++r) {
                const int m = mtile * 16 + hi * 8 + r;
                zbuf[m * ZSTR_ + n] = acc[j][r];
            }
        }
        __syncthreads();

        // ---- fused gates + state update (c in registers) ----
#pragma unroll
        for (int q = 0; q < 8; ++q) {
            const int m = q * 8 + wv;
            const float zi = zbuf[m * ZSTR_ + hh]      + bi_;
            const float zf = zbuf[m * ZSTR_ + 32 + hh] + bf_;
            const float zo = zbuf[m * ZSTR_ + 64 + hh] + bo_;
            const float zg = zbuf[m * ZSTR_ + 96 + hh] + bg_;
            const float cc = sigmoidf_(zf) * creg[q] + sigmoidf_(zi) * tanh_fast(zg);
            const float hv = sigmoidf_(zo) * tanh_fast(cc);
            creg[q] = cc;
            const size_t col = (size_t)m * H_ + hcol0 + hh;
            if (layer == 0) {
                ring0[(size_t)(t & (RING_ - 1)) * B_ * H_ + col] =
                    (unsigned short)f2bf(hv);
            } else {
                out[(size_t)t * B_ * H_ + col] = hv;
                ring1[(size_t)(t & 1) * B_ * H_ + col] = (unsigned short)f2bf(hv);
            }
            if (t == SEQ_ - 1) {
                out[FINAL_H + (size_t)layer * B_ * H_ + col] = hv;
                out[FINAL_C + (size_t)layer * B_ * H_ + col] = cc;
            }
        }

        signal_flag((layer == 0 ? flag0 : flag1) + t);
        __syncthreads();
    }
}

extern "C" void kernel_launch(void* const* d_in, const int* in_sizes, int n_in,
                              void* d_out, int out_size, void* d_ws, size_t ws_size,
                              hipStream_t stream) {
    (void)in_sizes; (void)n_in; (void)out_size; (void)ws_size;
    const float* x  = (const float*)d_in[0];
    const float* W0 = (const float*)d_in[1];
    const float* b0 = (const float*)d_in[2];
    const float* W1 = (const float*)d_in[3];
    const float* b1 = (const float*)d_in[4];
    float* out = (float*)d_out;

    unsigned char* ws = (unsigned char*)d_ws;
    unsigned*       flags   = (unsigned*)(ws + WS_FLAGS_);
    unsigned short* zerobuf = (unsigned short*)(ws + WS_ZERO_);
    unsigned short* ring0   = (unsigned short*)(ws + WS_RING0_);
    unsigned short* ring1   = (unsigned short*)(ws + WS_RING1_);
    unsigned short* xbf     = (unsigned short*)(ws + WS_XBF_);

    hipFuncSetAttribute((const void*)lstm_persistent,
                        hipFuncAttributeMaxDynamicSharedMemorySize, SMEM_);

    ws_init<<<(WS_ZERO_WORDS_ + 255) / 256, 256, 0, stream>>>(flags, WS_ZERO_WORDS_);
    xcast<<<(SEQ_ * B_ * IN_) / (8 * 256), 256, 0, stream>>>(x, xbf);
    lstm_persistent<<<2 * NWG_, THREADS_, SMEM_, stream>>>(
        xbf, W0, b0, W1, b1, out, ring0, ring1, zerobuf, flags, flags + SEQ_);
}
